// Attention_5738076307482
// MI455X (gfx1250) — matrix operand order repaired and checked by reading, not compiled, not run on hardware
//
#include <hip/hip_runtime.h>

#define BB 64
#define SS 1024
#define XD 256
#define YD 128

typedef __attribute__((ext_vector_type(16))) __bf16          v16bf;
typedef __attribute__((ext_vector_type(8)))  float           v8f;
typedef __attribute__((ext_vector_type(8)))  unsigned short  u16x8;
typedef unsigned short ushort_t;

// ---- packed f32->bf16 (RNE) : 2 elems -> 1 dword, final pack is one v_perm_b32
__device__ inline unsigned pkbf(float a, float b) {
  unsigned ua = __float_as_uint(a), ub = __float_as_uint(b);
  ua += 0x7FFFu + ((ua >> 16) & 1u);
  ub += 0x7FFFu + ((ub >> 16) & 1u);
  // result[15:0] = ua[31:16], result[31:16] = ub[31:16]
  return __builtin_amdgcn_perm(ub, ua, 0x07060302u);
}

__device__ inline ushort_t f2bf(float f) {
  unsigned u = __float_as_uint(f);
  return (ushort_t)((u + 0x7FFFu + ((u >> 16) & 1u)) >> 16);
}

__device__ inline unsigned encf(float f) {
  unsigned u = __float_as_uint(f);
  return (u & 0x80000000u) ? ~u : (u | 0x80000000u);
}
__device__ inline float decf(unsigned e) {
  unsigned u = (e & 0x80000000u) ? (e & 0x7FFFFFFFu) : ~e;
  return __uint_as_float(u);
}

__device__ inline v8f wmma_bf16(v16bf a, v16bf b, v8f c) {
  return __builtin_amdgcn_wmma_f32_16x16x32_bf16(false, a, false, b, (short)0, c,
                                                 false, false);
}

// A-matrix 16x32 bf16 layout (contiguous, matches B):
// lanes 0-15 : e0..15 -> K=e
// lanes 16-31: e0..15 -> K=16+e
__device__ inline v16bf load_a_bf(const ushort_t* row32, int hi) {
  const u16x8* p = (const u16x8*)row32;   // 64B-aligned 32-elem chunk
  union { v16bf v; u16x8 h[2]; } r;
  r.h[0] = p[2 * hi];
  r.h[1] = p[2 * hi + 1];
  return r.v;
}

// B-matrix 32x16 bf16: lane n holds column n&15; element e -> K = e + 16*(lane>=16)
__device__ inline v16bf load_b_bf_contig(const ushort_t* p) {  // 16 contiguous, 32B aligned
  const u16x8* q = (const u16x8*)p;
  union { v16bf v; u16x8 h[2]; } r;
  r.h[0] = q[0];
  r.h[1] = q[1];
  return r.v;
}

__device__ inline v16bf load_b_lds_strided(const ushort_t* base) {  // stride = YD elems
  union { v16bf v; ushort_t u[16]; } r;
#pragma unroll
  for (int e = 0; e < 16; ++e) r.u[e] = base[e * YD];
  return r.v;
}

// ---- CDNA5 async global->LDS path (signature confirmed by compiler probe) ----
#if __has_builtin(__builtin_amdgcn_global_load_async_to_lds_b128)
#define HAVE_ASYNC_TO_LDS 1
typedef int v4i __attribute__((__vector_size__(16)));
typedef __attribute__((address_space(1))) v4i* gv4i_p;   // global int4*
typedef __attribute__((address_space(3))) v4i* lv4i_p;   // lds int4*
__device__ inline void async_copy16(const void* g, void* l) {
  __builtin_amdgcn_global_load_async_to_lds_b128(
      (gv4i_p)(unsigned long long)g,
      (lv4i_p)(unsigned)(unsigned long long)l, 0, 0);
}
__device__ inline void wait_async_zero() {
#if __has_builtin(__builtin_amdgcn_s_wait_asynccnt)
  __builtin_amdgcn_s_wait_asynccnt(0);
#else
  asm volatile("s_wait_asynccnt 0x0" ::: "memory");
#endif
}
#endif

// ---------------------------------------------------------------- kernel: f32 -> bf16
__global__ void k_cvt(const float* __restrict__ s, ushort_t* __restrict__ d, int n8) {
  int i = blockIdx.x * blockDim.x + threadIdx.x;
  if (i >= n8) return;
  const float4* p = (const float4*)(s + (size_t)i * 8);
  float4 a = p[0];
  float4 b = p[1];
  uint4 o;
  o.x = pkbf(a.x, a.y);
  o.y = pkbf(a.z, a.w);
  o.z = pkbf(b.x, b.y);
  o.w = pkbf(b.z, b.w);
  *(uint4*)(d + (size_t)i * 8) = o;
}

// ---------------------------------------------------------------- kernel 0
__global__ void k_init_out(unsigned* __restrict__ outEnc) {
  int i = blockIdx.x * blockDim.x + threadIdx.x;
  if (i < BB * YD) outEnc[i] = 0x007FFFFFu;  // encf(-inf)
}

// ---------------------------------------------------------------- kernel 1
// Q/K/V = q @ W^T + b (all-bf16 operands). One wave: 1 m-tile x all 8 n-tiles of one
// matrix -> 64 WMMAs per wave, A fragment reused 8x per k-chunk.
__global__ void k_proj(const ushort_t* __restrict__ qbf, const ushort_t* __restrict__ Wbf,
                       const float* __restrict__ bq, const float* __restrict__ bk,
                       const float* __restrict__ bv,
                       ushort_t* __restrict__ Qbf, ushort_t* __restrict__ Kbf,
                       ushort_t* __restrict__ Vbf) {
  int wave = (blockIdx.x * blockDim.x + threadIdx.x) >> 5;
  int lane = threadIdx.x & 31;
  int mat = wave % 3;
  int mt  = wave / 3;
  if (mt >= (BB * SS) / 16) return;

  const ushort_t* W    = Wbf + (size_t)mat * YD * XD;
  const float*    bias = (mat == 0) ? bq : (mat == 1) ? bk : bv;
  ushort_t*       Out  = (mat == 0) ? Qbf : (mat == 1) ? Kbf : Vbf;

  int hi = lane >> 4;
  int ln = lane & 15;
  const ushort_t* arow = qbf + (size_t)(mt * 16 + ln) * XD;

  v8f acc[8];
#pragma unroll
  for (int nt = 0; nt < 8; ++nt)
#pragma unroll
    for (int v = 0; v < 8; ++v) acc[nt][v] = 0.f;

#pragma unroll
  for (int kk = 0; kk < XD; kk += 32) {
    v16bf A = load_a_bf(arow + kk, hi);
#pragma unroll
    for (int nt = 0; nt < 8; ++nt) {
      v16bf Bf = load_b_bf_contig(W + (size_t)(nt * 16 + ln) * XD + kk + 16 * hi);
      acc[nt] = wmma_bf16(A, Bf, acc[nt]);
    }
  }

#pragma unroll
  for (int nt = 0; nt < 8; ++nt) {
    float bc = bias[nt * 16 + ln];
#pragma unroll
    for (int v = 0; v < 8; ++v) {
      int row = mt * 16 + v + 8 * hi;
      Out[(size_t)row * YD + nt * 16 + ln] = f2bf(acc[nt][v] + bc);
    }
  }
}

// ---------------------------------------------------------------- kernel 2
// per-column softmax stats over q axis: m[b,k], c[b,k]; K-row fragments hoisted.
__global__ void k_stats(const ushort_t* __restrict__ Qbf, const ushort_t* __restrict__ Kbf,
                        float* __restrict__ mcol, float* __restrict__ ccol) {
  int wave = (blockIdx.x * blockDim.x + threadIdx.x) >> 5;
  int lane = threadIdx.x & 31;
  int b  = wave >> 6;
  int kt = wave & 63;
  int hi = lane >> 4, ln = lane & 15;
  const float scale = 0.08838834764831845f;  // 1/sqrt(128)

  const ushort_t* Qb = Qbf + (size_t)b * SS * YD;
  int kcol = kt * 16 + ln;
  const ushort_t* brow = Kbf + (size_t)b * SS * YD + (size_t)kcol * YD;

  v16bf Bfrag[4];
#pragma unroll
  for (int d = 0; d < 4; ++d) Bfrag[d] = load_b_bf_contig(brow + d * 32 + 16 * hi);

  float mrun = -3.0e38f, crun = 0.f;
  for (int qt = 0; qt < 64; ++qt) {
    const ushort_t* arow = Qb + (size_t)(qt * 16 + ln) * YD;
    v8f sc = {0.f, 0.f, 0.f, 0.f, 0.f, 0.f, 0.f, 0.f};
#pragma unroll
    for (int d = 0; d < 4; ++d)
      sc = wmma_bf16(load_a_bf(arow + d * 32, hi), Bfrag[d], sc);

    float tmax = -3.0e38f;
#pragma unroll
    for (int v = 0; v < 8; ++v) { sc[v] *= scale; tmax = fmaxf(tmax, sc[v]); }
    tmax = fmaxf(tmax, __shfl_xor(tmax, 16, 32));  // lanes n / n+16 share a column
    float mnew = fmaxf(mrun, tmax);
    float s = 0.f;
#pragma unroll
    for (int v = 0; v < 8; ++v) s += __expf(sc[v] - mnew);
    s += __shfl_xor(s, 16, 32);
    crun = crun * __expf(mrun - mnew) + s;
    mrun = mnew;
  }
  if (lane < 16) {
    mcol[b * SS + kcol] = mrun;
    ccol[b * SS + kcol] = crun;
  }
}

// ---------------------------------------------------------------- kernel 3
// out tile = sum_k exp(s-m[k])/c[k] * V[k,:]; V chunk staged to LDS (async),
// E' tile shared via LDS; column-max over q -> atomicMax(d_out).
__global__ void k_attn(const ushort_t* __restrict__ Qbf, const ushort_t* __restrict__ Kbf,
                       const ushort_t* __restrict__ Vbf,
                       const float* __restrict__ mcol, const float* __restrict__ ccol,
                       unsigned* __restrict__ outEnc) {
  __shared__ ushort_t Vlds[128][YD];   // 32 KB: V chunk (128 k rows x 128 d)
  __shared__ ushort_t Ebuf[16][128];   // 4 KB : E' tile (16 q x 128 k)

  int wv = threadIdx.x >> 5;   // 0..7
  int lane = threadIdx.x & 31;
  int hi = lane >> 4, ln = lane & 15;
  int b  = blockIdx.x >> 6;
  int qt = blockIdx.x & 63;
  const float scale = 0.08838834764831845f;

  const ushort_t* Qb = Qbf + (size_t)b * SS * YD;
  const ushort_t* Kb = Kbf + (size_t)b * SS * YD;
  const ushort_t* Vb = Vbf + (size_t)b * SS * YD;

  // Q-tile A fragments are loop-invariant: hoist.
  const ushort_t* arow = Qb + (size_t)(qt * 16 + ln) * YD;
  v16bf Afrag[4];
#pragma unroll
  for (int d = 0; d < 4; ++d) Afrag[d] = load_a_bf(arow + d * 32, hi);

  v8f acc = {0.f, 0.f, 0.f, 0.f, 0.f, 0.f, 0.f, 0.f};
  for (int kc = 0; kc < 8; ++kc) {
    // ---- stage V[kc*128 .. +128)[0..128) into LDS (16B per transfer, coalesced)
    {
      const char* gsrc = (const char*)(Vb + (size_t)kc * 128 * YD);
      char* ldst = (char*)&Vlds[0][0];
#pragma unroll
      for (int j = 0; j < 8; ++j) {
        int u = (threadIdx.x + j * 256) * 16;
#ifdef HAVE_ASYNC_TO_LDS
        async_copy16(gsrc + u, ldst + u);
#else
        *(uint4*)(ldst + u) = *(const uint4*)(gsrc + u);
#endif
      }
    }

    // ---- phase 1 (overlaps the V copy): this wave's 16 k-columns of E'
    int kbase = kc * 128 + wv * 16;
    const ushort_t* brow = Kb + (size_t)(kbase + ln) * YD;
    v8f sc = {0.f, 0.f, 0.f, 0.f, 0.f, 0.f, 0.f, 0.f};
#pragma unroll
    for (int d = 0; d < 4; ++d)
      sc = wmma_bf16(Afrag[d], load_b_bf_contig(brow + d * 32 + 16 * hi), sc);

    float mv  = mcol[b * SS + kbase + ln];
    float inv = 1.f / ccol[b * SS + kbase + ln];
#pragma unroll
    for (int v = 0; v < 8; ++v)
      Ebuf[v + 8 * hi][wv * 16 + ln] = f2bf(__expf(sc[v] * scale - mv) * inv);

#ifdef HAVE_ASYNC_TO_LDS
    wait_async_zero();
#endif
    __syncthreads();

    // ---- phase 2: acc += E'(16x128, LDS) @ V(128x16, LDS) for this wave's d-block
    int d = wv * 16 + ln;
#pragma unroll
    for (int kk = 0; kk < 128; kk += 32) {
      v16bf A = load_a_bf(&Ebuf[ln][kk], hi);
      acc = wmma_bf16(A, load_b_lds_strided(&Vlds[kk + 16 * hi][d]), acc);
    }
    __syncthreads();
  }

  float t = acc[0];
#pragma unroll
  for (int v = 1; v < 8; ++v) t = fmaxf(t, acc[v]);
  t = fmaxf(t, __shfl_xor(t, 16, 32));
  if (lane < 16) atomicMax(&outEnc[b * YD + wv * 16 + ln], encf(t));
}

// ---------------------------------------------------------------- kernel 4
__global__ void k_decode(unsigned* __restrict__ enc, float* __restrict__ out) {
  int i = blockIdx.x * blockDim.x + threadIdx.x;
  if (i < BB * YD) out[i] = decf(enc[i]);
}

// ----------------------------------------------------------------
extern "C" void kernel_launch(void* const* d_in, const int* in_sizes, int n_in,
                              void* d_out, int out_size, void* d_ws, size_t ws_size,
                              hipStream_t stream) {
  (void)in_sizes; (void)n_in; (void)out_size; (void)ws_size;
  const float* q  = (const float*)d_in[0];
  const float* Wq = (const float*)d_in[1];
  const float* bq = (const float*)d_in[2];
  const float* Wk = (const float*)d_in[3];
  const float* bk = (const float*)d_in[4];
  const float* Wv = (const float*)d_in[5];
  const float* bv = (const float*)d_in[6];

  char* ws = (char*)d_ws;
  size_t off = 0;
  const size_t qkvBytes = (size_t)BB * SS * YD * sizeof(ushort_t);  // 16 MB each
  ushort_t* Qbf = (ushort_t*)(ws + off); off += qkvBytes;
  ushort_t* Kbf = (ushort_t*)(ws + off); off += qkvBytes;
  ushort_t* Vbf = (ushort_t*)(ws + off); off += qkvBytes;
  ushort_t* qbf = (ushort_t*)(ws + off); off += (size_t)BB * SS * XD * sizeof(ushort_t);
  ushort_t* Wbf = (ushort_t*)(ws + off); off += (size_t)3 * YD * XD * sizeof(ushort_t);
  float* mcol = (float*)(ws + off); off += (size_t)BB * SS * sizeof(float);
  float* ccol = (float*)(ws + off); off += (size_t)BB * SS * sizeof(float);

  unsigned* outEnc = (unsigned*)d_out;

  // pre-convert activations + weights to bf16 (packed v_perm_b32 path)
  {
    int n8 = (BB * SS * XD) / 8;
    k_cvt<<<(n8 + 255) / 256, 256, 0, stream>>>(q, qbf, n8);
    int w8 = (YD * XD) / 8;
    k_cvt<<<(w8 + 255) / 256, 256, 0, stream>>>(Wq, Wbf + 0 * YD * XD, w8);
    k_cvt<<<(w8 + 255) / 256, 256, 0, stream>>>(Wk, Wbf + 1 * YD * XD, w8);
    k_cvt<<<(w8 + 255) / 256, 256, 0, stream>>>(Wv, Wbf + 2 * YD * XD, w8);
  }

  k_init_out<<<(BB * YD + 255) / 256, 256, 0, stream>>>(outEnc);

  // proj: 4096 m-tiles * 3 matrices = 12288 waves, 8 waves/block
  k_proj<<<12288 / 8, 256, 0, stream>>>(qbf, Wbf, bq, bk, bv, Qbf, Kbf, Vbf);

  // stats: 64 batches * 64 k-tiles = 4096 waves
  k_stats<<<4096 / 8, 256, 0, stream>>>(Qbf, Kbf, mcol, ccol);

  // attn: one block per (batch, q-tile)
  k_attn<<<BB * (SS / 16), 256, 0, stream>>>(Qbf, Kbf, Vbf, mcol, ccol, outEnc);

  k_decode<<<(BB * YD + 255) / 256, 256, 0, stream>>>(outEnc, (float*)d_out);
}
